// AttentionModule_59176059404750
// MI455X (gfx1250) — matrix operand order repaired and checked by reading, not compiled, not run on hardware
//
#include <hip/hip_runtime.h>

// ---------------------------------------------------------------------------
// Problem constants (reference: B=8, C=N=O=2048)
// ---------------------------------------------------------------------------
#define DIM 2048
#define NBATCH 8

#define TILE_M 128
#define TILE_N 128
#define TILE_K 32
#define LDSS   40   // LDS row stride in ushort (32 + 8 pad) = 80B, multiple of 16B

typedef __attribute__((ext_vector_type(2)))  __bf16 v2bf;
typedef __attribute__((ext_vector_type(8)))  __bf16 v8bf;
typedef __attribute__((ext_vector_type(16))) __bf16 v16bf;
typedef __attribute__((ext_vector_type(8)))  float  v8f;

// Native BF16 converts (RTNE) — lowers to v_cvt_*bf16_f32 on gfx1250.
__device__ __forceinline__ unsigned short f2bf(float f) {
  __bf16 h = (__bf16)f;
  return __builtin_bit_cast(unsigned short, h);
}
__device__ __forceinline__ unsigned int pack2(float a, float b) {
  v2bf v = { (__bf16)a, (__bf16)b };
  return __builtin_bit_cast(unsigned int, v);
}

// Assemble a 16-element bf16 fragment from two 16B LDS chunks.
__device__ __forceinline__ v16bf frag_from(const unsigned short* p0,
                                           const unsigned short* p1) {
  v8bf lo = *(const v8bf*)p0;
  v8bf hi = *(const v8bf*)p1;
  return __builtin_shufflevector(lo, hi, 0,1,2,3,4,5,6,7,8,9,10,11,12,13,14,15);
}

// ---------------------------------------------------------------------------
// Templated WMMA GEMM: D = A(MxK) * B(KxN) (+ epilogue)
//   A_F32 / B_F32 : operand dtype in global memory (f32 converted to bf16 on
//                   the fly while staging to LDS; else raw bf16/ushort copy)
//   B_TRANS       : B given as (N x K) row-major, i.e. B[k][n] = Bg[n][k]
//   EPI: 0 = bf16 store of (acc + bias[row])          (projections -> Q/K/V)
//        1 = f32 store of acc                         (attention logits S)
//        2 = f32 store of gamma*acc + residual[idx]   (final output)
// Workgroup: 256 threads = 8 waves; macro tile 128x128; wave tile 32x64
// (2x4 grid of v_wmma_f32_16x16x32_bf16 accumulators).
// ---------------------------------------------------------------------------
template<int A_F32, int B_F32, int B_TRANS, int EPI>
__global__ __launch_bounds__(256)
void gemm_wmma(const void* __restrict__ Ag, const void* __restrict__ Bg,
               const float* __restrict__ bias,
               unsigned short* __restrict__ Dbf, float* __restrict__ Df,
               const float* __restrict__ residual,
               const float* __restrict__ gscale,
               int M, int N, int K)
{
  __shared__ __align__(16) unsigned short As[TILE_M * LDSS];
  __shared__ __align__(16) unsigned short Bs[TILE_N * LDSS];

  const int tid   = threadIdx.x;
  const int lane  = tid & 31;
  const int wave  = tid >> 5;
  const int waveM = wave & 3;   // 4 waves along M -> 32 rows each
  const int waveN = wave >> 2;  // 2 waves along N -> 64 cols each
  const int g     = lane >> 4;  // lane half (K-group select)
  const int lm    = lane & 15;
  const int blockM = blockIdx.y * TILE_M;
  const int blockN = blockIdx.x * TILE_N;

  const v8f vzero = {0.f,0.f,0.f,0.f,0.f,0.f,0.f,0.f};
  v8f acc[2][4];
  #pragma unroll
  for (int mt = 0; mt < 2; ++mt)
    #pragma unroll
    for (int nt = 0; nt < 4; ++nt) acc[mt][nt] = vzero;

  for (int k0 = 0; k0 < K; k0 += TILE_K) {
    // ---- stage A tile: 128 rows x 32 K, As[row][k]
    {
      const int row = tid >> 1;
      const int kh  = (tid & 1) << 4;   // 0 or 16
      unsigned short* dst = &As[row * LDSS + kh];
      if constexpr (A_F32) {
        const float4* s4 =
            (const float4*)((const float*)Ag + (size_t)(blockM + row) * K + k0 + kh);
        float4 f0 = s4[0], f1 = s4[1], f2 = s4[2], f3 = s4[3];
        uint4 u0, u1;
        u0.x = pack2(f0.x, f0.y); u0.y = pack2(f0.z, f0.w);
        u0.z = pack2(f1.x, f1.y); u0.w = pack2(f1.z, f1.w);
        u1.x = pack2(f2.x, f2.y); u1.y = pack2(f2.z, f2.w);
        u1.z = pack2(f3.x, f3.y); u1.w = pack2(f3.z, f3.w);
        *(uint4*)dst = u0;
        *(uint4*)(dst + 8) = u1;
      } else {
        const uint4* s4 =
            (const uint4*)((const unsigned short*)Ag + (size_t)(blockM + row) * K + k0 + kh);
        *(uint4*)dst = s4[0];
        *(uint4*)(dst + 8) = s4[1];
      }
      if (k0 + TILE_K < K) {  // prefetch next K tile of A (global_prefetch_b8)
        const size_t eb = (A_F32 ? 4 : 2);
        __builtin_prefetch((const char*)Ag +
            ((size_t)(blockM + row) * K + k0 + TILE_K + kh) * eb, 0, 1);
      }
    }
    // ---- stage B tile as Bs[n][k] (so frag reads are contiguous per lane)
    if constexpr (B_TRANS) {
      const int n  = tid >> 1;
      const int kh = (tid & 1) << 4;
      unsigned short* dst = &Bs[n * LDSS + kh];
      if constexpr (B_F32) {
        const float4* s4 =
            (const float4*)((const float*)Bg + (size_t)(blockN + n) * K + k0 + kh);
        float4 f0 = s4[0], f1 = s4[1], f2 = s4[2], f3 = s4[3];
        uint4 u0, u1;
        u0.x = pack2(f0.x, f0.y); u0.y = pack2(f0.z, f0.w);
        u0.z = pack2(f1.x, f1.y); u0.w = pack2(f1.z, f1.w);
        u1.x = pack2(f2.x, f2.y); u1.y = pack2(f2.z, f2.w);
        u1.z = pack2(f3.x, f3.y); u1.w = pack2(f3.z, f3.w);
        *(uint4*)dst = u0;
        *(uint4*)(dst + 8) = u1;
      } else {
        const uint4* s4 =
            (const uint4*)((const unsigned short*)Bg + (size_t)(blockN + n) * K + k0 + kh);
        *(uint4*)dst = s4[0];
        *(uint4*)(dst + 8) = s4[1];
      }
    } else {
      const int k  = tid & 31;          // one K-row per thread
      const int n0 = (tid >> 5) << 4;   // 16-wide N segment
      if constexpr (B_F32) {
        const float4* s4 =
            (const float4*)((const float*)Bg + (size_t)(k0 + k) * N + blockN + n0);
        float tmp[16];
        ((float4*)tmp)[0] = s4[0]; ((float4*)tmp)[1] = s4[1];
        ((float4*)tmp)[2] = s4[2]; ((float4*)tmp)[3] = s4[3];
        #pragma unroll
        for (int e = 0; e < 16; ++e)
          Bs[(n0 + e) * LDSS + k] = f2bf(tmp[e]);
      } else {
        unsigned int w[8];
        const uint4* s4 =
            (const uint4*)((const unsigned short*)Bg + (size_t)(k0 + k) * N + blockN + n0);
        ((uint4*)w)[0] = s4[0]; ((uint4*)w)[1] = s4[1];
        #pragma unroll
        for (int e = 0; e < 8; ++e) {
          Bs[(n0 + 2*e    ) * LDSS + k] = (unsigned short)(w[e] & 0xffffu);
          Bs[(n0 + 2*e + 1) * LDSS + k] = (unsigned short)(w[e] >> 16);
        }
      }
    }
    __syncthreads();

    // ---- fragments per the CDNA5 16-bit WMMA VGPR layouts
    v16bf afrag[2], bfrag[4];
    #pragma unroll
    for (int mt = 0; mt < 2; ++mt) {
      const int r = waveM * 32 + mt * 16 + lm;
      // A 16x32: lane half g holds K = [g*8, g*8+8) and [16+g*8, 16+g*8+8)
      afrag[mt] = frag_from(&As[r * LDSS + g * 8],
                            &As[r * LDSS + 16 + g * 8]);
    }
    #pragma unroll
    for (int nt = 0; nt < 4; ++nt) {
      const int c = waveN * 64 + nt * 16 + lm;
      // B 32x16: lane half g holds K = [g*8, g*8+8) and [16+g*8, 16+g*8+8) for column c
      bfrag[nt] = frag_from(&Bs[c * LDSS + g * 8],
                            &Bs[c * LDSS + 16 + g * 8]);
    }
    #pragma unroll
    for (int mt = 0; mt < 2; ++mt)
      #pragma unroll
      for (int nt = 0; nt < 4; ++nt)
        acc[mt][nt] = __builtin_amdgcn_wmma_f32_16x16x32_bf16(
            false, afrag[mt], false, bfrag[nt],
            (short)0, acc[mt][nt], false, false);
    __syncthreads();
  }

  // ---- epilogue: C/D layout = VGPR r -> row (g*8 + r), col = lane&15
  float gm = 0.f;
  if constexpr (EPI == 2) gm = gscale[0];
  #pragma unroll
  for (int mt = 0; mt < 2; ++mt) {
    #pragma unroll
    for (int nt = 0; nt < 4; ++nt) {
      const int rowb = blockM + waveM * 32 + mt * 16 + g * 8;
      const int col  = blockN + waveN * 64 + nt * 16 + lm;
      #pragma unroll
      for (int r = 0; r < 8; ++r) {
        const size_t idx = (size_t)(rowb + r) * N + col;
        const float v = acc[mt][nt][r];
        if constexpr (EPI == 0) {
          Dbf[idx] = f2bf(v + bias[rowb + r]);
        } else if constexpr (EPI == 1) {
          Df[idx] = v;
        } else {
          Df[idx] = gm * v + residual[idx];
        }
      }
    }
  }
}

// ---------------------------------------------------------------------------
// Row softmax: S (f32, row-major N x N) -> P (bf16)
// ---------------------------------------------------------------------------
__global__ __launch_bounds__(256)
void softmax_rows(const float* __restrict__ S, unsigned short* __restrict__ P,
                  int n)
{
  __shared__ float red[256];
  const int row = blockIdx.x;
  const int tid = threadIdx.x;
  const float* sp = S + (size_t)row * n;

  float m = -3.4e38f;
  for (int j = tid; j < n; j += 256) m = fmaxf(m, sp[j]);
  red[tid] = m; __syncthreads();
  #pragma unroll
  for (int s = 128; s > 0; s >>= 1) {
    if (tid < s) red[tid] = fmaxf(red[tid], red[tid + s]);
    __syncthreads();
  }
  const float rmax = red[0]; __syncthreads();

  float sum = 0.f;
  for (int j = tid; j < n; j += 256) sum += __expf(sp[j] - rmax);
  red[tid] = sum; __syncthreads();
  #pragma unroll
  for (int s = 128; s > 0; s >>= 1) {
    if (tid < s) red[tid] += red[tid + s];
    __syncthreads();
  }
  const float inv = 1.0f / red[0];

  unsigned short* pp = P + (size_t)row * n;
  for (int j = tid; j < n; j += 256)
    pp[j] = f2bf(__expf(sp[j] - rmax) * inv);
}

// ---------------------------------------------------------------------------
// Host launcher. Workspace layout (per-batch reuse, ~48 MB total):
//   Qb, Kb, Vb, Pb : 2048*2048 bf16 each (8 MB each)
//   S              : 2048*2048 f32 (16 MB)
// ---------------------------------------------------------------------------
extern "C" void kernel_launch(void* const* d_in, const int* in_sizes, int n_in,
                              void* d_out, int out_size, void* d_ws, size_t ws_size,
                              hipStream_t stream) {
  (void)in_sizes; (void)n_in; (void)out_size; (void)ws_size;

  const float* x     = (const float*)d_in[0];
  const float* Wq    = (const float*)d_in[1];
  const float* bq    = (const float*)d_in[2];
  const float* Wk    = (const float*)d_in[3];
  const float* bk    = (const float*)d_in[4];
  const float* Wv    = (const float*)d_in[5];
  const float* bv    = (const float*)d_in[6];
  const float* gamma = (const float*)d_in[7];
  float* out = (float*)d_out;

  const size_t MAT = (size_t)DIM * DIM;
  unsigned short* Qb = (unsigned short*)d_ws;
  unsigned short* Kb = Qb + MAT;
  unsigned short* Vb = Kb + MAT;
  unsigned short* Pb = Vb + MAT;
  float*          S  = (float*)(Pb + MAT);

  dim3 grid(DIM / TILE_N, DIM / TILE_M);
  dim3 block(256);

  for (int b = 0; b < NBATCH; ++b) {
    const float* xb = x + (size_t)b * MAT;

    // Projections: Q/K/V = W * X + bias  (f32 in, bf16 out)
    gemm_wmma<1,1,0,0><<<grid, block, 0, stream>>>(
        Wq, xb, bq, Qb, nullptr, nullptr, nullptr, DIM, DIM, DIM);
    gemm_wmma<1,1,0,0><<<grid, block, 0, stream>>>(
        Wk, xb, bk, Kb, nullptr, nullptr, nullptr, DIM, DIM, DIM);
    gemm_wmma<1,1,0,0><<<grid, block, 0, stream>>>(
        Wv, xb, bv, Vb, nullptr, nullptr, nullptr, DIM, DIM, DIM);

    // S = Q * K^T  (bf16 in, f32 out)
    gemm_wmma<0,0,1,1><<<grid, block, 0, stream>>>(
        Qb, Kb, nullptr, nullptr, S, nullptr, nullptr, DIM, DIM, DIM);

    // P = softmax_rows(S)  (bf16 out)
    softmax_rows<<<DIM, block, 0, stream>>>(S, Pb, DIM);

    // out = gamma * (P * V) + x
    gemm_wmma<0,0,0,2><<<grid, block, 0, stream>>>(
        Pb, Vb, nullptr, nullptr, out + (size_t)b * MAT, xb, gamma,
        DIM, DIM, DIM);
  }
}
